// Network_79809082294541
// MI455X (gfx1250) — matrix operand order repaired and checked by reading, not compiled, not run on hardware
//
#include <hip/hip_runtime.h>
#include <hip/hip_bf16.h>
#include <stdint.h>

typedef __attribute__((ext_vector_type(16))) _Float16 v16h;
typedef __attribute__((ext_vector_type(8)))  _Float16 v8h;
typedef __attribute__((ext_vector_type(8)))  float    v8f;

static inline int cdiv(int a, int b) { return (a + b - 1) / b; }
__device__ __forceinline__ int iclamp(int v, int lo, int hi) {
    return v < lo ? lo : (v > hi ? hi : v);
}

// ---------------------------------------------------------------------------
// Weight prepack: f32 OIHW [Cout][Cin][3][3]  ->  f16 [9][Coutp][Cinp]
// (zero padded so WMMA K/M tiles are exact)
// ---------------------------------------------------------------------------
__global__ void prepack_w_k(const float* __restrict__ w, _Float16* __restrict__ wpk,
                            int Cout, int Cin, int Coutp, int Cinp) {
    int idx = blockIdx.x * blockDim.x + threadIdx.x;
    int total = 9 * Coutp * Cinp;
    if (idx >= total) return;
    int ci = idx % Cinp;
    int t  = idx / Cinp;
    int co = t % Coutp;
    int rs = t / Coutp;
    float v = 0.0f;
    if (co < Cout && ci < Cin) v = w[(size_t)(co * Cin + ci) * 9 + rs];
    wpk[idx] = (_Float16)v;
}

// ---------------------------------------------------------------------------
// 3x3 conv (pad=1) as implicit GEMM on v_wmma_f32_16x16x32_f16.
// Block = 128 threads = 4 waves. Wave w computes a 16(Cout) x 32(x-pixels)
// output tile (two 16x16 WMMA tiles sharing one A fragment) on row
// (blockIdx.y*4 + w). One LDS stage of [6row][34col][32cin] f16 serves all
// 9 taps of a 32-wide K chunk -> 18 WMMAs per stage, A fetched once per tap.
// ---------------------------------------------------------------------------
__global__ __launch_bounds__(128)
void conv3x3_wmma_k(const float* __restrict__ in, const _Float16* __restrict__ wpk,
                    const float* __restrict__ bias, const float* __restrict__ skip,
                    float* __restrict__ out,
                    int Cin, int Cinp, int Cout, int Coutp,
                    int H, int W, int relu) {
    __shared__ _Float16 lds[6 * 34 * 32];     // [pos = ry*34+rx][ci] ; 13056 B

    const int tid  = threadIdx.x;
    const int lane = tid & 31;
    const int wave = tid >> 5;
    const int px   = lane & 15;               // N index (pixel within tile)
    const int hi   = lane >> 4;               // half-wave select
    const int xbase  = blockIdx.x * 32;
    const int ybase  = blockIdx.y * 4;
    const int cobase = blockIdx.z * 16;
    const int HWl = H * W;
    const int y = ybase + wave;

    v8f acc0 = {};
    v8f acc1 = {};

    for (int cc = 0; cc < Cinp; cc += 32) {
        __syncthreads();
        // ---- stage input slab (rows ybase-1..ybase+4, cols xbase-1..xbase+32)
        for (int t = tid; t < 6 * 34 * 32; t += 128) {
            int ci = t / 204;                 // 0..31
            int p  = t % 204;                 // 0..203
            int ry = p / 34, rx = p % 34;
            int gy = ybase - 1 + ry;
            int gx = xbase - 1 + rx;
            int c  = cc + ci;
            float v = 0.0f;
            if (c < Cin && gy >= 0 && gy < H && gx >= 0 && gx < W)
                v = in[(size_t)c * HWl + gy * W + gx];
            lds[p * 32 + ci] = (_Float16)v;
        }
        __syncthreads();

        // ---- 9 taps, two WMMAs each (A reused across both N tiles)
#pragma unroll
        for (int r = 0; r < 3; ++r) {
#pragma unroll
            for (int s = 0; s < 3; ++s) {
                const int rs = r * 3 + s;
                // A fragment (weights) straight from global (L2-hot):
                // halves 0..7  = K(cc + 8*hi .. +7), halves 8..15 = +16
                const _Float16* wl =
                    wpk + ((size_t)rs * Coutp + (cobase + px)) * Cinp + cc + hi * 8;
                v8h alo = *(const v8h*)(wl);
                v8h ahi = *(const v8h*)(wl + 16);
                v16h a = __builtin_shufflevector(alo, ahi,
                          0,1,2,3,4,5,6,7,8,9,10,11,12,13,14,15);
                // B fragments from LDS: halves 0..7 = K(cc + 8*hi .. +7), halves 8..15 = +16, N = px
                const int ry = wave + r;
                const int pos0 = ry * 34 + px + s;
                const _Float16* bl0 = &lds[pos0 * 32 + hi * 8];
                v8h b0lo = *(const v8h*)(bl0);
                v8h b0hi = *(const v8h*)(bl0 + 16);
                v16h b0 = __builtin_shufflevector(b0lo, b0hi,
                          0,1,2,3,4,5,6,7,8,9,10,11,12,13,14,15);
                acc0 = __builtin_amdgcn_wmma_f32_16x16x32_f16(
                          false, a, false, b0, (short)0, acc0, false, false);
                const _Float16* bl1 = &lds[(pos0 + 16) * 32 + hi * 8];
                v8h b1lo = *(const v8h*)(bl1);
                v8h b1hi = *(const v8h*)(bl1 + 16);
                v16h b1 = __builtin_shufflevector(b1lo, b1hi,
                          0,1,2,3,4,5,6,7,8,9,10,11,12,13,14,15);
                acc1 = __builtin_amdgcn_wmma_f32_16x16x32_f16(
                          false, a, false, b1, (short)0, acc1, false, false);
            }
        }
    }

    // ---- epilogue: bias (+ReLU) (+skip), masked stores for both N tiles
    if (y < H) {
        const int x0 = xbase + px;
        const int x1 = x0 + 16;
#pragma unroll
        for (int j = 0; j < 8; ++j) {
            int m  = j + hi * 8;              // C/D layout: VGPR j -> M row
            int co = cobase + m;
            if (co < Cout) {
                float bv = bias[co];
                size_t rowoff = (size_t)co * HWl + (size_t)y * W;
                if (x0 < W) {
                    float v = acc0[j] + bv;
                    if (relu) v = fmaxf(v, 0.0f);
                    if (skip) v += skip[rowoff + x0];
                    out[rowoff + x0] = v;
                }
                if (x1 < W) {
                    float v = acc1[j] + bv;
                    if (relu) v = fmaxf(v, 0.0f);
                    if (skip) v += skip[rowoff + x1];
                    out[rowoff + x1] = v;
                }
            }
        }
    }
}

// ---------------------------------------------------------------------------
// Elementwise helpers
// ---------------------------------------------------------------------------
__global__ void avgpool2_k(const float* __restrict__ in, float* __restrict__ out,
                           int C, int H, int W) {
    int oh = H >> 1, ow = W >> 1;
    int idx = blockIdx.x * blockDim.x + threadIdx.x;
    if (idx >= C * oh * ow) return;
    int x = idx % ow; int t = idx / ow; int y = t % oh; int c = t / oh;
    const float* p = in + ((size_t)c * H + 2 * y) * W + 2 * x;
    out[idx] = 0.25f * (p[0] + p[1] + p[W] + p[W + 1]);
}

__global__ void resize_bilinear_k(const float* __restrict__ in, float* __restrict__ out,
                                  int C, int IH, int IW, int OH, int OW) {
    int idx = blockIdx.x * blockDim.x + threadIdx.x;
    if (idx >= C * OH * OW) return;
    int ox = idx % OW; int t = idx / OW; int oy = t % OH; int c = t / OH;
    float sy = (OH > 1) ? (float)(IH - 1) / (float)(OH - 1) : 0.0f;
    float sx = (OW > 1) ? (float)(IW - 1) / (float)(OW - 1) : 0.0f;
    float fy = oy * sy, fx = ox * sx;
    int y0 = (int)floorf(fy), x0 = (int)floorf(fx);
    int y1 = min(y0 + 1, IH - 1), x1 = min(x0 + 1, IW - 1);
    float wy = fy - (float)y0, wx = fx - (float)x0;
    const float* p = in + (size_t)c * IH * IW;
    float a = p[(size_t)y0 * IW + x0] * (1.0f - wx) + p[(size_t)y0 * IW + x1] * wx;
    float b = p[(size_t)y1 * IW + x0] * (1.0f - wx) + p[(size_t)y1 * IW + x1] * wx;
    out[idx] = a * (1.0f - wy) + b * wy;
}

// adaptive separable conv, replicate padding folded in via clamped indexing
__global__ void sepconv_k(const float* __restrict__ in, const float* __restrict__ vv,
                          const float* __restrict__ hh, const float* __restrict__ res,
                          float* __restrict__ out, int C, int H, int W, int K) {
    int idx = blockIdx.x * blockDim.x + threadIdx.x;
    int HWl = H * W;
    if (idx >= C * HWl) return;
    int x = idx % W; int t = idx / W; int y = t % H; int c = t / H;
    int pad = (K - 1) >> 1;
    const float* pin = in + (size_t)c * HWl;
    int pix = y * W + x;
    float acc = 0.0f;
    for (int i = 0; i < K; ++i) {
        int yy = iclamp(y + i - pad, 0, H - 1);
        const float* pr = pin + (size_t)yy * W;
        float row = 0.0f;
        for (int j = 0; j < K; ++j) {
            int xx = iclamp(x + j - pad, 0, W - 1);
            row += pr[xx] * hh[(size_t)j * HWl + pix];
        }
        acc += row * vv[(size_t)i * HWl + pix];
    }
    if (res) acc += res[idx];
    out[idx] = acc;
}

__global__ void wcombine_k(const float* __restrict__ d1, const float* __restrict__ d2,
                           const float* __restrict__ w2, float* __restrict__ out,
                           int C, int HWl) {
    int idx = blockIdx.x * blockDim.x + threadIdx.x;
    if (idx >= C * HWl) return;
    int p = idx % HWl;
    out[idx] = d1[idx] * w2[p] + d2[idx] * w2[HWl + p];
}

// ---------------------------------------------------------------------------
// Host orchestration
// ---------------------------------------------------------------------------
extern "C" void kernel_launch(void* const* d_in, const int* in_sizes, int n_in,
                              void* d_out, int out_size, void* d_ws, size_t ws_size,
                              hipStream_t stream) {
    (void)in_sizes; (void)n_in; (void)out_size; (void)ws_size;

    const int H = 256, W = 384;
    const int H2 = 128, W2 = 192, H4 = 64, W4 = 96, H8 = 32, W8 = 48;
    const int H16 = 16, W16 = 24, H32 = 8, W32 = 12;
    const size_t HW  = (size_t)H * W;     // 98304
    const size_t hw2 = (size_t)H2 * W2;   // 24576
    const size_t hw4 = (size_t)H4 * W4;   // 6144
    const size_t hw8 = (size_t)H8 * W8;   // 1536
    const size_t hw16 = (size_t)H16 * W16;
    const size_t hw32 = (size_t)H32 * W32;

    const float* ti1 = (const float*)d_in[0];
    const float* ti2 = (const float*)d_in[1];
    const float* m1  = (const float*)d_in[2];
    const float* m2  = (const float*)d_in[3];
    // (w,b) pairs flattened in params-dict insertion order, starting at d_in[4]
    auto PW = [&](int k) { return (const float*)d_in[4 + 2 * k]; };
    auto PB = [&](int k) { return (const float*)d_in[5 + 2 * k]; };

    float* wsf = (float*)d_ws;
    size_t off = 0;
    auto alloc = [&](size_t n) {
        float* p = wsf + off;
        off += (n + 63) & ~(size_t)63;
        return p;
    };

    // reusable f16 weight-pack buffer, sized for the largest layer (512x512)
    _Float16* wpk = (_Float16*)alloc((size_t)9 * 512 * 512 / 2 + 64);

    // ------------------- primitive launchers -------------------
    auto resize = [&](const float* in, float* out, int C, int ih, int iw, int oh, int ow) {
        int tot = C * oh * ow;
        resize_bilinear_k<<<cdiv(tot, 256), 256, 0, stream>>>(in, out, C, ih, iw, oh, ow);
    };
    auto avgpool = [&](const float* in, float* out, int C, int h, int w) {
        int tot = C * (h / 2) * (w / 2);
        avgpool2_k<<<cdiv(tot, 256), 256, 0, stream>>>(in, out, C, h, w);
    };
    auto sepconv = [&](const float* in, const float* v, const float* h, const float* res,
                       float* out, int C, int hh_, int ww_, int K) {
        int tot = C * hh_ * ww_;
        sepconv_k<<<cdiv(tot, 256), 256, 0, stream>>>(in, v, h, res, out, C, hh_, ww_, K);
    };
    auto wcomb = [&](const float* d1, const float* d2, const float* w2, float* out,
                     int C, size_t plane) {
        int tot = C * (int)plane;
        wcombine_k<<<cdiv(tot, 256), 256, 0, stream>>>(d1, d2, w2, out, C, (int)plane);
    };
    auto conv = [&](const float* in, int k, const float* skip, float* out,
                    int Cin, int Cout, int h, int w, int relu) {
        int Cinp  = cdiv(Cin, 32) * 32;
        int Coutp = cdiv(Cout, 16) * 16;
        int tot = 9 * Coutp * Cinp;
        prepack_w_k<<<cdiv(tot, 256), 256, 0, stream>>>(PW(k), wpk, Cout, Cin, Coutp, Cinp);
        dim3 g(cdiv(w, 32), cdiv(h, 4), Coutp / 16);
        conv3x3_wmma_k<<<g, 128, 0, stream>>>(in, wpk, PB(k), skip, out,
                                              Cin, Cinp, Cout, Coutp, h, w, relu);
    };
    auto basic3 = [&](const float* x, int k0, float* pa, float* pb,
                      int Cin, int Cout, int h, int w) -> float* {
        conv(x,  k0 + 0, nullptr, pa, Cin,  Cout, h, w, 1);
        conv(pa, k0 + 1, nullptr, pb, Cout, Cout, h, w, 1);
        conv(pb, k0 + 2, nullptr, pa, Cout, Cout, h, w, 1);
        return pa;
    };
    auto subnetL = [&](const float* x, int k0, int Cin, int F, int h, int w,
                       float* pp, float* pq, float* fs, float* fu, float* out) {
        conv(x,  k0 + 0, nullptr, pp, Cin, Cin, h, w, 1);
        conv(pp, k0 + 1, nullptr, pq, Cin, Cin, h, w, 1);
        conv(pq, k0 + 2, nullptr, fs, Cin, F,   h, w, 1);
        resize(fs, fu, F, h, w, 2 * h, 2 * w);
        conv(fu, k0 + 3, nullptr, out, F, F, 2 * h, 2 * w, 0);   // no relu
    };

    // ------------------- encoder -------------------
    float* xcat = alloc(6 * HW);
    hipMemcpyAsync(xcat,           ti1, 3 * HW * sizeof(float), hipMemcpyDeviceToDevice, stream);
    hipMemcpyAsync(xcat + 3 * HW,  ti2, 3 * HW * sizeof(float), hipMemcpyDeviceToDevice, stream);

    float* a1 = alloc(32 * HW);  float* b1 = alloc(32 * HW);
    float* c1 = basic3(xcat, 0, a1, b1, 6, 32, H, W);
    float* p1 = alloc(32 * hw2); avgpool(c1, p1, 32, H, W);

    float* a2 = alloc(64 * hw2); float* b2 = alloc(64 * hw2);
    float* c2 = basic3(p1, 3, a2, b2, 32, 64, H2, W2);
    float* p2 = alloc(64 * hw4); avgpool(c2, p2, 64, H2, W2);

    float* a3 = alloc(128 * hw4); float* b3 = alloc(128 * hw4);
    float* c3 = basic3(p2, 6, a3, b3, 64, 128, H4, W4);
    float* p3 = alloc(128 * hw8); avgpool(c3, p3, 128, H4, W4);

    float* a4 = alloc(256 * hw8); float* b4 = alloc(256 * hw8);
    float* c4 = basic3(p3, 9, a4, b4, 128, 256, H8, W8);
    float* p4 = alloc(256 * hw16); avgpool(c4, p4, 256, H8, W8);

    float* a5 = alloc(512 * hw16); float* b5 = alloc(512 * hw16);
    float* c5 = basic3(p4, 12, a5, b5, 256, 512, H16, W16);
    float* p5 = alloc(512 * hw32); avgpool(c5, p5, 512, H16, W16);

    // ------------------- decoder top -------------------
    float* e5a = alloc(512 * hw32); float* e5b = alloc(512 * hw32);
    float* d5 = basic3(p5, 15, e5a, e5b, 512, 512, H32, W32);
    float* u5 = alloc(512 * hw16); resize(d5, u5, 512, H32, W32, H16, W16);
    float* comb5 = alloc(512 * hw16); conv(u5, 18, c5, comb5, 512, 512, H16, W16, 1);

    float* e4a = alloc(256 * hw16); float* e4b = alloc(256 * hw16);
    float* d4 = basic3(comb5, 19, e4a, e4b, 512, 256, H16, W16);
    float* u4 = alloc(256 * hw8); resize(d4, u4, 256, H16, W16, H8, W8);
    float* comb4 = alloc(256 * hw8); conv(u4, 22, c4, comb4, 256, 256, H8, W8, 1);

    // ------------------- level A (K=13, outputs at H/4) -------------------
    float* spa = alloc(258 * hw8); float* sqa = alloc(258 * hw8);
    float* fsa = alloc(13 * hw8);  float* fua = alloc(13 * hw4);
    float* va  = alloc(13 * hw4);  float* ha  = alloc(13 * hw4);
    float* in1q = alloc(3 * hw4);  resize(ti1, in1q, 3, H, W, H4, W4);
    float* in2q = alloc(3 * hw4);  resize(ti2, in2q, 3, H, W, H4, W4);
    float* dot1a = alloc(3 * hw4); float* dot2a = alloc(3 * hw4);

    subnetL(comb4, 23, 256, 13, H8, W8, spa, sqa, fsa, fua, va);   // mv1_a
    subnetL(comb4, 31, 256, 13, H8, W8, spa, sqa, fsa, fua, ha);   // mh1_a
    sepconv(in1q, va, ha, nullptr, dot1a, 3, H4, W4, 13);
    subnetL(comb4, 27, 256, 13, H8, W8, spa, sqa, fsa, fua, va);   // mv2_a
    subnetL(comb4, 35, 256, 13, H8, W8, spa, sqa, fsa, fua, ha);   // mh2_a
    sepconv(in2q, va, ha, nullptr, dot2a, 3, H4, W4, 13);

    float* wcat_a = alloc(258 * hw8);
    hipMemcpyAsync(wcat_a, comb4, 256 * hw8 * sizeof(float), hipMemcpyDeviceToDevice, stream);
    resize(m1, wcat_a + 256 * hw8, 1, H, W, H8, W8);
    resize(m2, wcat_a + 257 * hw8, 1, H, W, H8, W8);
    float* wa = alloc(2 * hw4);
    subnetL(wcat_a, 39, 258, 2, H8, W8, spa, sqa, fsa, fua, wa);   // w_a1
    float* res_a = alloc(3 * hw4);
    wcomb(dot1a, dot2a, wa, res_a, 3, hw4);

    // ------------------- level B (K=25, outputs at H/2) -------------------
    float* e3a = alloc(128 * hw8); float* e3b = alloc(128 * hw8);
    float* d3 = basic3(comb4, 43, e3a, e3b, 256, 128, H8, W8);
    float* u3 = alloc(128 * hw4); resize(d3, u3, 128, H8, W8, H4, W4);
    float* comb3 = alloc(128 * hw4); conv(u3, 46, c3, comb3, 128, 128, H4, W4, 1);

    float* spb = alloc(130 * hw4); float* sqb = alloc(130 * hw4);
    float* fsb = alloc(25 * hw4);  float* fub = alloc(25 * hw2);
    float* vb = alloc(25 * hw2);   float* hb = alloc(25 * hw2);
    float* in1h = alloc(3 * hw2);  resize(ti1, in1h, 3, H, W, H2, W2);
    float* in2h = alloc(3 * hw2);  resize(ti2, in2h, 3, H, W, H2, W2);
    float* resa_up = alloc(3 * hw2); resize(res_a, resa_up, 3, H4, W4, H2, W2);
    float* dot1b = alloc(3 * hw2); float* dot2b = alloc(3 * hw2);

    subnetL(comb3, 47, 128, 25, H4, W4, spb, sqb, fsb, fub, vb);   // mv1_b
    subnetL(comb3, 55, 128, 25, H4, W4, spb, sqb, fsb, fub, hb);   // mh1_b
    sepconv(in1h, vb, hb, resa_up, dot1b, 3, H2, W2, 25);
    subnetL(comb3, 51, 128, 25, H4, W4, spb, sqb, fsb, fub, vb);   // mv2_b
    subnetL(comb3, 59, 128, 25, H4, W4, spb, sqb, fsb, fub, hb);   // mh2_b
    sepconv(in2h, vb, hb, resa_up, dot2b, 3, H2, W2, 25);

    float* wcat_b = alloc(130 * hw4);
    hipMemcpyAsync(wcat_b, comb3, 128 * hw4 * sizeof(float), hipMemcpyDeviceToDevice, stream);
    resize(m1, wcat_b + 128 * hw4, 1, H, W, H4, W4);
    resize(m2, wcat_b + 129 * hw4, 1, H, W, H4, W4);
    float* wbv = alloc(2 * hw2);
    subnetL(wcat_b, 63, 130, 2, H4, W4, spb, sqb, fsb, fub, wbv);  // w_b1
    float* res_b = alloc(3 * hw2);
    wcomb(dot1b, dot2b, wbv, res_b, 3, hw2);

    // ------------------- level final (K=51, full res) -------------------
    float* e2a = alloc(64 * hw4); float* e2b = alloc(64 * hw4);
    float* d2 = basic3(comb3, 67, e2a, e2b, 128, 64, H4, W4);
    float* u2 = alloc(64 * hw2); resize(d2, u2, 64, H4, W4, H2, W2);
    float* comb2 = alloc(64 * hw2); conv(u2, 70, c2, comb2, 64, 64, H2, W2, 1);

    float* spc = alloc(66 * hw2); float* sqc = alloc(66 * hw2);
    float* fsc = alloc(51 * hw2); float* fuc = alloc(51 * HW);
    float* vc = alloc(51 * HW);   float* hc = alloc(51 * HW);
    float* resb_up = alloc(3 * HW); resize(res_b, resb_up, 3, H2, W2, H, W);
    float* dot1 = alloc(3 * HW);  float* dot2 = alloc(3 * HW);

    subnetL(comb2, 71, 64, 51, H2, W2, spc, sqc, fsc, fuc, vc);    // v1
    subnetL(comb2, 79, 64, 51, H2, W2, spc, sqc, fsc, fuc, hc);    // h1
    sepconv(ti1, vc, hc, resb_up, dot1, 3, H, W, 51);
    subnetL(comb2, 75, 64, 51, H2, W2, spc, sqc, fsc, fuc, vc);    // v2
    subnetL(comb2, 83, 64, 51, H2, W2, spc, sqc, fsc, fuc, hc);    // h2
    sepconv(ti2, vc, hc, resb_up, dot2, 3, H, W, 51);

    float* wcat_c = alloc(66 * hw2);
    hipMemcpyAsync(wcat_c, comb2, 64 * hw2 * sizeof(float), hipMemcpyDeviceToDevice, stream);
    resize(m1, wcat_c + 64 * hw2, 1, H, W, H2, W2);
    resize(m2, wcat_c + 65 * hw2, 1, H, W, H2, W2);
    float* wfin = alloc(2 * HW);
    subnetL(wcat_c, 87, 66, 2, H2, W2, spc, sqc, fsc, fuc, wfin);  // w1

    wcomb(dot1, dot2, wfin, (float*)d_out, 3, HW);
}
